// FedMoEModel_82231443849388
// MI455X (gfx1250) — matrix operand order repaired and checked by reading, not compiled, not run on hardware
//
#include <hip/hip_runtime.h>
#include <stdint.h>
#include <stddef.h>

// ---------------------------------------------------------------------------
// Types for CDNA5 WMMA (wave32): V_WMMA_F32_16X16X32_BF16
// ---------------------------------------------------------------------------
typedef __attribute__((ext_vector_type(16))) __bf16 v16bf;
typedef __attribute__((ext_vector_type(8)))  float  v8f;

union Frag  { v16bf v; uint4 q[2]; };

__device__ __forceinline__ unsigned short f2bf(float x) {
    unsigned int u = __float_as_uint(x);
    u += 0x7fffu + ((u >> 16) & 1u);           // round-to-nearest-even
    return (unsigned short)(u >> 16);
}
__device__ __forceinline__ float bf2f(unsigned short h) {
    return __uint_as_float(((unsigned int)h) << 16);
}

// gfx1250 async global->LDS copy (ASYNCcnt-tracked, no VGPR round trip).
__device__ __forceinline__ void async_ld_b128(unsigned lds_off, const void* gptr) {
    asm volatile("global_load_async_to_lds_b128 %0, %1, off"
                 :: "v"(lds_off), "v"((unsigned long long)(uintptr_t)gptr)
                 : "memory");
}
__device__ __forceinline__ void wait_async0() {
    asm volatile("s_wait_asynccnt 0" ::: "memory");
}

// ---------------------------------------------------------------------------
// Elementwise helpers
// ---------------------------------------------------------------------------
__global__ void k_cvt(const float* __restrict__ in, unsigned short* __restrict__ out, long n) {
    long i  = (long)blockIdx.x * 256 + threadIdx.x;
    long st = (long)gridDim.x * 256;
    for (; i < n; i += st) out[i] = f2bf(in[i]);
}

// fp32 [cnt][K][N] -> bf16 [cnt][N][K]  (one-time weight transpose)
__global__ void k_cvtT(const float* __restrict__ in, unsigned short* __restrict__ out,
                       int K, int N, long total) {
    long i  = (long)blockIdx.x * 256 + threadIdx.x;
    long st = (long)gridDim.x * 256;
    long KN = (long)K * N;
    for (; i < total; i += st) {
        long e = i / KN, r = i - e * KN;
        int k = (int)(r / N), n = (int)(r - (long)k * N);
        out[e * KN + (long)n * K + k] = f2bf(in[i]);
    }
}

__global__ void k_zero(float* __restrict__ p, long n) {
    long i  = (long)blockIdx.x * 256 + threadIdx.x;
    long st = (long)gridDim.x * 256;
    for (; i < n; i += st) p[i] = 0.f;
}

// Fold eval-mode BN (+ preceding linear bias) into per-channel scale/shift.
__global__ void k_fold(const float* __restrict__ g,  const float* __restrict__ bb,
                       const float* __restrict__ rm, const float* __restrict__ rv,
                       const float* __restrict__ lb, float* __restrict__ s,
                       float* __restrict__ t, int n) {
    int i = blockIdx.x * 256 + threadIdx.x;
    if (i < n) {
        float sc = g[i] * rsqrtf(rv[i] + 1e-5f);
        s[i] = sc;
        float l = lb ? lb[i] : 0.f;
        t[i] = (l - rm[i]) * sc + bb[i];
    }
}

// ---------------------------------------------------------------------------
// bf16 GEMM with WMMA:  C[M,N] = A[M,K] * Bt[N,K]^T  (+ templated epilogue)
// Block tile 128x64, 256 threads = 8 waves, wave tile 32x32 (2x2 frags),
// k-step 64 (two WMMA sub-steps), DOUBLE-BUFFERED LDS staged with
// global_load_async_to_lds_b128 so HBM traffic overlaps WMMA.
// Requires M%128==0, N%64==0, K%64==0, lda multiple of 8.
// ---------------------------------------------------------------------------
template<bool SC, bool BI, bool RL, bool RW, bool OF, bool AC, bool OH>
__global__ __launch_bounds__(256) void k_gemm(
    const unsigned short* __restrict__ A, int lda,
    const unsigned short* __restrict__ Bt, int K,
    const float* __restrict__ scale, const float* __restrict__ bias,
    const float* __restrict__ roww, int rws,
    float* __restrict__ outF, unsigned short* __restrict__ outH, int ldo)
{
    __shared__ __align__(16) unsigned short lA[2 * 128 * 64];   // 32 KB ping-pong
    __shared__ __align__(16) unsigned short lB[2 * 64 * 64];    // 16 KB ping-pong

    const int tid  = threadIdx.x;
    const int lane = tid & 31;
    const int wave = tid >> 5;
    const int m0   = blockIdx.y * 128;
    const int n0   = blockIdx.x * 64;
    const int mw   = (wave & 3) * 32;
    const int nw   = (wave >> 2) * 32;

    v8f acc[2][2];
#pragma unroll
    for (int i = 0; i < 2; i++)
#pragma unroll
        for (int j = 0; j < 2; j++)
#pragma unroll
            for (int r = 0; r < 8; r++) acc[i][j][r] = 0.f;

    const int arow = tid >> 1, aoff = (tid & 1) * 32;   // halves within A row
    const int brow = tid >> 2, boff = (tid & 3) * 16;   // halves within B row
    const unsigned ldsA = (unsigned)(uintptr_t)&lA[0];
    const unsigned ldsB = (unsigned)(uintptr_t)&lB[0];
    const int ml  = lane & 15;
    const int kbA = (lane < 16) ? 0 : 8;                // A frag K grouping (ISA 7.12.2)
    const int kbB = (lane < 16) ? 0 : 8;                // B frag K grouping

    auto stage = [&](int k0, int buf) {
        const unsigned short* ga = A + (size_t)(m0 + arow) * lda + k0 + aoff;
        unsigned baseA = ldsA + (unsigned)buf * (128 * 64 * 2);
#pragma unroll
        for (int j = 0; j < 4; j++)
            async_ld_b128(baseA + (unsigned)(arow * 64 + aoff + j * 8) * 2, ga + j * 8);
        const unsigned short* gb = Bt + (size_t)(n0 + brow) * K + k0 + boff;
        unsigned baseB = ldsB + (unsigned)buf * (64 * 64 * 2);
#pragma unroll
        for (int j = 0; j < 2; j++)
            async_ld_b128(baseB + (unsigned)(brow * 64 + boff + j * 8) * 2, gb + j * 8);
    };

    const int nt = K >> 6;
    stage(0, 0);                                        // prologue
    for (int t = 0; t < nt; t++) {
        wait_async0();          // this wave's tile-t copies done
        __syncthreads();        // tile t resident; prior reads of other buf done
        if (t + 1 < nt) stage((t + 1) * 64, (t + 1) & 1);   // overlap with compute

        const unsigned short* cA = &lA[(t & 1) * (128 * 64)];
        const unsigned short* cB = &lB[(t & 1) * (64 * 64)];
#pragma unroll
        for (int s = 0; s < 2; s++) {                   // two K=32 sub-steps
            Frag af[2], bfr[2];
#pragma unroll
            for (int fm = 0; fm < 2; fm++) {
                const unsigned short* p = cA + (mw + fm * 16 + ml) * 64 + s * 32 + kbA;
                af[fm].q[0] = *(const uint4*)p;         // K = kbA .. kbA+7
                af[fm].q[1] = *(const uint4*)(p + 16);  // K = kbA+16 .. kbA+23
            }
#pragma unroll
            for (int fn = 0; fn < 2; fn++) {
                const unsigned short* p = cB + (nw + fn * 16 + ml) * 64 + s * 32 + kbB;
                bfr[fn].q[0] = *(const uint4*)p;        // K = kbB .. kbB+7
                bfr[fn].q[1] = *(const uint4*)(p + 16); // K = kbB+16 .. kbB+23
            }
#pragma unroll
            for (int fm = 0; fm < 2; fm++)
#pragma unroll
                for (int fn = 0; fn < 2; fn++)
                    acc[fm][fn] = __builtin_amdgcn_wmma_f32_16x16x32_bf16(
                        false, af[fm].v, false, bfr[fn].v, (short)0, acc[fm][fn], false, false);
        }
    }

    const int rsel = (lane >> 4) * 8;  // C layout: M = r + (lane>=16 ? 8 : 0)
#pragma unroll
    for (int fm = 0; fm < 2; fm++)
#pragma unroll
        for (int fn = 0; fn < 2; fn++)
#pragma unroll
            for (int r = 0; r < 8; r++) {
                int row = m0 + mw + fm * 16 + r + rsel;
                int col = n0 + nw + fn * 16 + ml;
                float v = acc[fm][fn][r];
                if (SC)      v = v * scale[col] + bias[col];
                else if (BI) v += bias[col];
                if (RL)      v = fmaxf(v, 0.f);
                if (RW)      v *= roww[(size_t)row * rws];
                size_t o = (size_t)row * ldo + col;
                if (OF) { if (AC) outF[o] += v; else outF[o] = v; }
                if (OH) outH[o] = f2bf(v);
            }
}

// ---------------------------------------------------------------------------
// conv1 3x3 (1->32) + BN + relu + 2x2 maxpool, direct VALU (K=9 too small for
// WMMA). out: [B][16][16][32] bf16.
// ---------------------------------------------------------------------------
__global__ void k_conv1(const float* __restrict__ img, const float* __restrict__ w,
                        const float* __restrict__ scale, const float* __restrict__ bias,
                        unsigned short* __restrict__ out, int Bn)
{
    int idx = blockIdx.x * 256 + threadIdx.x;
    if (idx >= Bn * (16 * 16 * 32)) return;
    int c  = idx & 31;
    int px = (idx >> 5) & 15;
    int py = (idx >> 9) & 15;
    int b  = idx >> 13;
    float sc = scale[c], tc = bias[c];
    float mx = -1e30f;
#pragma unroll
    for (int dy = 0; dy < 2; dy++)
#pragma unroll
        for (int dx = 0; dx < 2; dx++) {
            int oy = py * 2 + dy, ox = px * 2 + dx;
            float s = 0.f;
#pragma unroll
            for (int ky = 0; ky < 3; ky++)
#pragma unroll
                for (int kx = 0; kx < 3; kx++) {
                    int iy = oy + ky - 1, ix = ox + kx - 1;
                    if (iy >= 0 && iy < 32 && ix >= 0 && ix < 32)
                        s += img[((size_t)b * 32 + iy) * 32 + ix] * w[(ky * 3 + kx) * 32 + c];
                }
            mx = fmaxf(mx, fmaxf(s * sc + tc, 0.f));
        }
    out[idx] = f2bf(mx);
}

// ---------------------------------------------------------------------------
// conv2 3x3 (32->64) + BN + relu + 2x2 maxpool, WMMA implicit-im2col.
// One block per sample: M=256 pixels, N=64, K=288 (9 k-steps; step s == tap).
// Weights pre-transposed [64][288] in global -> straight async copy to LDS.
// ---------------------------------------------------------------------------
__global__ __launch_bounds__(256) void k_conv2(
    const unsigned short* __restrict__ in,   // [B][16][16][32] bf16
    const unsigned short* __restrict__ wT,   // [64][288] bf16
    const float* __restrict__ scale, const float* __restrict__ bias,
    unsigned short* __restrict__ out)        // [B][8][8][64] bf16
{
    __shared__ __align__(16) unsigned short lIn[18 * 18 * 32];
    __shared__ __align__(16) unsigned short lW[64 * 288];   // reused as [256][64]

    const int tid  = threadIdx.x;
    const int lane = tid & 31;
    const int wave = tid >> 5;
    const int b    = blockIdx.x;
    const unsigned ldsW  = (unsigned)(uintptr_t)&lW[0];
    const unsigned ldsIn = (unsigned)(uintptr_t)&lIn[0];

    // weights: 64*288 halves = 2304 x b128 chunks, 9 per thread (async)
#pragma unroll
    for (int j = 0; j < 9; j++) {
        int i = tid + j * 256;
        async_ld_b128(ldsW + (unsigned)i * 16, wT + (size_t)i * 8);
    }
    // padded input: zero border (normal DS stores, disjoint from async interior)
    for (int i = tid; i < 18 * 18 * 32; i += 256) {
        int sp = i >> 5;
        int x = sp % 18, y = sp / 18;
        if (x == 0 || x == 17 || y == 0 || y == 17) lIn[i] = 0;
    }
    // interior rows: 16 rows x 512 halves = 1024 x b128 chunks, 4 per thread
#pragma unroll
    for (int j = 0; j < 4; j++) {
        int i   = tid + j * 256;
        int row = i >> 6;            // 64 chunks per row
        int cr  = i & 63;
        unsigned lhalf = (unsigned)(((row + 1) * 18 + 1) * 32 + cr * 8);
        async_ld_b128(ldsIn + lhalf * 2, in + (size_t)(b * 16 + row) * 512 + cr * 8);
    }
    wait_async0();
    __syncthreads();

    v8f acc[2][4];
#pragma unroll
    for (int i = 0; i < 2; i++)
#pragma unroll
        for (int j = 0; j < 4; j++)
#pragma unroll
            for (int r = 0; r < 8; r++) acc[i][j][r] = 0.f;

    const int pbase = wave * 32;
    const int ml    = lane & 15;
    const int kbA   = (lane < 16) ? 0 : 8;
    const int kbB   = (lane < 16) ? 0 : 8;

#pragma unroll
    for (int s9 = 0; s9 < 9; s9++) {
        const int ky = s9 / 3, kx = s9 % 3;
        Frag af[2], bfr[4];
#pragma unroll
        for (int fm = 0; fm < 2; fm++) {
            int p = pbase + fm * 16 + ml;
            int py = p >> 4, px = p & 15;
            const unsigned short* q = &lIn[((py + ky) * 18 + (px + kx)) * 32 + kbA];
            af[fm].q[0] = *(const uint4*)q;
            af[fm].q[1] = *(const uint4*)(q + 16);
        }
#pragma unroll
        for (int fn = 0; fn < 4; fn++) {
            const unsigned short* q = &lW[(fn * 16 + ml) * 288 + s9 * 32 + kbB];
            bfr[fn].q[0] = *(const uint4*)q;
            bfr[fn].q[1] = *(const uint4*)(q + 16);
        }
#pragma unroll
        for (int fm = 0; fm < 2; fm++)
#pragma unroll
            for (int fn = 0; fn < 4; fn++)
                acc[fm][fn] = __builtin_amdgcn_wmma_f32_16x16x32_bf16(
                    false, af[fm].v, false, bfr[fn].v, (short)0, acc[fm][fn], false, false);
    }
    __syncthreads();   // all lW reads done; safe to reuse as conv output

    const int rsel = (lane >> 4) * 8;
#pragma unroll
    for (int fm = 0; fm < 2; fm++)
#pragma unroll
        for (int fn = 0; fn < 4; fn++)
#pragma unroll
            for (int r = 0; r < 8; r++) {
                int p  = pbase + fm * 16 + r + rsel;
                int ch = fn * 16 + ml;
                float v = fmaxf(acc[fm][fn][r] * scale[ch] + bias[ch], 0.f);
                lW[p * 64 + ch] = f2bf(v);
            }
    __syncthreads();

    for (int i = tid; i < 8 * 8 * 64; i += 256) {   // 2x2 maxpool
        int c = i & 63, qx = (i >> 6) & 7, qy = i >> 9;
        int p00 = (qy * 2) * 16 + qx * 2;
        float v0 = bf2f(lW[p00 * 64 + c]);
        float v1 = bf2f(lW[(p00 + 1) * 64 + c]);
        float v2 = bf2f(lW[(p00 + 16) * 64 + c]);
        float v3 = bf2f(lW[(p00 + 17) * 64 + c]);
        out[(size_t)b * 4096 + i] = f2bf(fmaxf(fmaxf(v0, v1), fmaxf(v2, v3)));
    }
}

// ---------------------------------------------------------------------------
// Router: logits(768->8), softmax probs, top-2, gate softmax, combine weights.
// ---------------------------------------------------------------------------
__global__ void k_gate(const unsigned short* __restrict__ fused,
                       const float* __restrict__ gw, const float* __restrict__ gb,
                       float* __restrict__ probs, float* __restrict__ comb, int Bn)
{
    int bi = blockIdx.x * blockDim.x + threadIdx.x;
    if (bi >= Bn) return;
    float lg[8];
#pragma unroll
    for (int e = 0; e < 8; e++) lg[e] = gb[e];
    const unsigned short* xr = fused + (size_t)bi * 768;
    for (int f = 0; f < 768; f++) {
        float xv = bf2f(xr[f]);
        const float* wr = gw + f * 8;
#pragma unroll
        for (int e = 0; e < 8; e++) lg[e] += xv * wr[e];
    }
    float mx = lg[0];
#pragma unroll
    for (int e = 1; e < 8; e++) mx = fmaxf(mx, lg[e]);
    float ex[8], s = 0.f;
#pragma unroll
    for (int e = 0; e < 8; e++) { ex[e] = expf(lg[e] - mx); s += ex[e]; }
#pragma unroll
    for (int e = 0; e < 8; e++) probs[bi * 8 + e] = ex[e] / s;

    int i1 = 0; float v1 = lg[0];
#pragma unroll
    for (int e = 1; e < 8; e++) if (lg[e] > v1) { v1 = lg[e]; i1 = e; }
    int i2 = -1; float v2 = -1e30f;
#pragma unroll
    for (int e = 0; e < 8; e++) if (e != i1 && lg[e] > v2) { v2 = lg[e]; i2 = e; }
    float e2 = expf(v2 - v1);
    float g1 = 1.f / (1.f + e2);
    float g2 = e2 / (1.f + e2);
#pragma unroll
    for (int e = 0; e < 8; e++) comb[bi * 8 + e] = 0.f;
    comb[bi * 8 + i1] = g1;
    comb[bi * 8 + i2] = g2;
}

// aux = 8 * sum_e importance_e * load_e ; fixed-order tree reduction.
__global__ void k_aux(const float* __restrict__ probs, const float* __restrict__ comb,
                      float* __restrict__ aux_out, int Bn)
{
    __shared__ float simp[256][8];
    __shared__ float sld[256][8];
    int tid = threadIdx.x;
    float imp[8], ld[8];
#pragma unroll
    for (int e = 0; e < 8; e++) { imp[e] = 0.f; ld[e] = 0.f; }
    for (int b = tid; b < Bn; b += 256)
#pragma unroll
        for (int e = 0; e < 8; e++) {
            imp[e] += probs[b * 8 + e];
            ld[e]  += (comb[b * 8 + e] > 0.f) ? 1.f : 0.f;
        }
#pragma unroll
    for (int e = 0; e < 8; e++) { simp[tid][e] = imp[e]; sld[tid][e] = ld[e]; }
    __syncthreads();
    for (int off = 128; off > 0; off >>= 1) {
        if (tid < off)
#pragma unroll
            for (int e = 0; e < 8; e++) {
                simp[tid][e] += simp[tid + off][e];
                sld[tid][e]  += sld[tid + off][e];
            }
        __syncthreads();
    }
    if (tid == 0) {
        float a = 0.f;
#pragma unroll
        for (int e = 0; e < 8; e++) a += simp[0][e] * sld[0][e];
        aux_out[0] = 8.f * a;
    }
}

// Final classifier layer (N=12, VALU): logits -> d_out fp32
__global__ void k_cls2(const unsigned short* __restrict__ h, const float* __restrict__ w,
                       const float* __restrict__ b2, float* __restrict__ out, int Bn)
{
    int idx = blockIdx.x * 256 + threadIdx.x;
    if (idx >= Bn * 12) return;
    int j = idx % 12, bb = idx / 12;
    float acc = b2[j];
    const unsigned short* hr = h + (size_t)bb * 128;
    for (int k = 0; k < 128; k++) acc += bf2f(hr[k]) * w[k * 12 + j];
    out[idx] = acc;
}

// ---------------------------------------------------------------------------
// Host orchestration
// ---------------------------------------------------------------------------
extern "C" void kernel_launch(void* const* d_in, const int* in_sizes, int n_in,
                              void* d_out, int out_size, void* d_ws, size_t ws_size,
                              hipStream_t stream)
{
    (void)in_sizes; (void)n_in; (void)out_size; (void)ws_size;
    const int B = 2048;

    const float* x_csv   = (const float*)d_in[0];
    const float* x_img1  = (const float*)d_in[1];
    const float* x_img2  = (const float*)d_in[2];
    const float* sen_w1  = (const float*)d_in[3];
    const float* sen_b1  = (const float*)d_in[4];
    const float* bn1_g   = (const float*)d_in[5];
    const float* bn1_b   = (const float*)d_in[6];
    const float* bn1_rm  = (const float*)d_in[7];
    const float* bn1_rv  = (const float*)d_in[8];
    const float* sen_w2  = (const float*)d_in[9];
    const float* sen_b2  = (const float*)d_in[10];
    const float* bn2_g   = (const float*)d_in[11];
    const float* bn2_b   = (const float*)d_in[12];
    const float* bn2_rm  = (const float*)d_in[13];
    const float* bn2_rv  = (const float*)d_in[14];
    const float* c1_w    = (const float*)d_in[15];
    const float* c1_b    = (const float*)d_in[16];
    const float* bnc1_g  = (const float*)d_in[17];
    const float* bnc1_b  = (const float*)d_in[18];
    const float* bnc1_rm = (const float*)d_in[19];
    const float* bnc1_rv = (const float*)d_in[20];
    const float* c2_w    = (const float*)d_in[21];
    const float* c2_b    = (const float*)d_in[22];
    const float* bnc2_g  = (const float*)d_in[23];
    const float* bnc2_b  = (const float*)d_in[24];
    const float* bnc2_rm = (const float*)d_in[25];
    const float* bnc2_rv = (const float*)d_in[26];
    const float* ifc_w   = (const float*)d_in[27];
    const float* ifc_b   = (const float*)d_in[28];
    const float* bni_g   = (const float*)d_in[29];
    const float* bni_b   = (const float*)d_in[30];
    const float* bni_rm  = (const float*)d_in[31];
    const float* bni_rv  = (const float*)d_in[32];
    const float* gate_w  = (const float*)d_in[33];
    const float* gate_b  = (const float*)d_in[34];
    const float* ew1     = (const float*)d_in[35];
    const float* eb1     = (const float*)d_in[36];
    const float* ew2     = (const float*)d_in[37];
    const float* eb2     = (const float*)d_in[38];
    const float* cl_w1   = (const float*)d_in[39];
    const float* cl_b1   = (const float*)d_in[40];
    const float* cl_w2   = (const float*)d_in[41];
    const float* cl_b2   = (const float*)d_in[42];

    size_t off = 0;
    auto alloc = [&](size_t bytes) -> void* {
        void* p = (char*)d_ws + off;
        off += (bytes + 255) & ~(size_t)255;
        return p;
    };
    unsigned short* sw1b  = (unsigned short*)alloc((size_t)512 * 512 * 2);
    unsigned short* sw2b  = (unsigned short*)alloc((size_t)512 * 256 * 2);
    unsigned short* c2wb  = (unsigned short*)alloc((size_t)288 * 64 * 2);
    unsigned short* ifcwb = (unsigned short*)alloc((size_t)4096 * 256 * 2);
    unsigned short* ew1b  = (unsigned short*)alloc((size_t)8 * 768 * 512 * 2);
    unsigned short* ew2b  = (unsigned short*)alloc((size_t)8 * 512 * 256 * 2);
    unsigned short* cw1b  = (unsigned short*)alloc((size_t)256 * 128 * 2);
    float* s1  = (float*)alloc(512 * 4);  float* t1  = (float*)alloc(512 * 4);
    float* s2  = (float*)alloc(256 * 4);  float* t2  = (float*)alloc(256 * 4);
    float* sc1 = (float*)alloc(32 * 4);   float* tc1 = (float*)alloc(32 * 4);
    float* sc2 = (float*)alloc(64 * 4);   float* tc2 = (float*)alloc(64 * 4);
    float* si  = (float*)alloc(256 * 4);  float* ti  = (float*)alloc(256 * 4);
    unsigned short* xcb    = (unsigned short*)alloc((size_t)B * 512 * 2);
    unsigned short* h1b    = (unsigned short*)alloc((size_t)B * 512 * 2);
    unsigned short* fusedb = (unsigned short*)alloc((size_t)B * 768 * 2);
    unsigned short* poolA  = (unsigned short*)alloc((size_t)B * 16 * 16 * 32 * 2);
    unsigned short* poolB  = (unsigned short*)alloc((size_t)B * 4096 * 2);
    unsigned short* heb    = (unsigned short*)alloc((size_t)B * 512 * 2);
    float*          accF   = (float*)alloc((size_t)B * 256 * 4);
    unsigned short* accb   = (unsigned short*)alloc((size_t)B * 256 * 2);
    unsigned short* hclb   = (unsigned short*)alloc((size_t)B * 128 * 2);
    float* probs = (float*)alloc((size_t)B * 8 * 4);
    float* comb  = (float*)alloc((size_t)B * 8 * 4);

    auto cvt = [&](const float* src, unsigned short* dst, long n) {
        int blocks = (int)((n + 255) / 256); if (blocks > 16384) blocks = 16384;
        k_cvt<<<dim3(blocks), dim3(256), 0, stream>>>(src, dst, n);
    };
    auto cvtT = [&](const float* src, unsigned short* dst, int K, int N, int cnt) {
        long total = (long)K * N * cnt;
        int blocks = (int)((total + 255) / 256); if (blocks > 16384) blocks = 16384;
        k_cvtT<<<dim3(blocks), dim3(256), 0, stream>>>(src, dst, K, N, total);
    };
    // GEMM variants (templated epilogues; no runtime branching in kernels)
    auto gemmBN = [&](const unsigned short* A, int lda, const unsigned short* Bt,
                      int M, int N, int K, const float* scale, const float* bias,
                      unsigned short* oH, int ldo) {   // scale/bias + relu -> bf16
        k_gemm<true, true, true, false, false, false, true>
            <<<dim3(N / 64, M / 128), dim3(256), 0, stream>>>(
                A, lda, Bt, K, scale, bias, nullptr, 0, nullptr, oH, ldo);
    };
    auto gemmB = [&](const unsigned short* A, int lda, const unsigned short* Bt,
                     int M, int N, int K, const float* bias,
                     unsigned short* oH, int ldo) {    // bias + relu -> bf16
        k_gemm<false, true, true, false, false, false, true>
            <<<dim3(N / 64, M / 128), dim3(256), 0, stream>>>(
                A, lda, Bt, K, nullptr, bias, nullptr, 0, nullptr, oH, ldo);
    };
    auto gemmEx2 = [&](const unsigned short* A, int lda, const unsigned short* Bt,
                       int M, int N, int K, const float* bias,
                       const float* roww, int rws, float* oF, int ldo) {
        // bias + relu, per-row gate weight, accumulate fp32
        k_gemm<false, true, true, true, true, true, false>
            <<<dim3(N / 64, M / 128), dim3(256), 0, stream>>>(
                A, lda, Bt, K, nullptr, bias, roww, rws, oF, nullptr, ldo);
    };

    // --- weight conversion + transpose, activation conversion ---
    cvtT(sen_w1, sw1b, 512, 512, 1);
    cvtT(sen_w2, sw2b, 512, 256, 1);
    cvtT(c2_w,   c2wb, 288, 64, 1);
    cvtT(ifc_w,  ifcwb, 4096, 256, 1);
    cvtT(ew1,    ew1b, 768, 512, 8);
    cvtT(ew2,    ew2b, 512, 256, 8);
    cvtT(cl_w1,  cw1b, 256, 128, 1);
    cvt(x_csv, xcb, (long)B * 512);

    // --- BN folds (linear/conv bias folded in) ---
    k_fold<<<dim3(2), dim3(256), 0, stream>>>(bn1_g, bn1_b, bn1_rm, bn1_rv, sen_b1, s1, t1, 512);
    k_fold<<<dim3(1), dim3(256), 0, stream>>>(bn2_g, bn2_b, bn2_rm, bn2_rv, sen_b2, s2, t2, 256);
    k_fold<<<dim3(1), dim3(256), 0, stream>>>(bnc1_g, bnc1_b, bnc1_rm, bnc1_rv, c1_b, sc1, tc1, 32);
    k_fold<<<dim3(1), dim3(256), 0, stream>>>(bnc2_g, bnc2_b, bnc2_rm, bnc2_rv, c2_b, sc2, tc2, 64);
    k_fold<<<dim3(1), dim3(256), 0, stream>>>(bni_g, bni_b, bni_rm, bni_rv, ifc_b, si, ti, 256);

    // --- sensor encoder ---
    gemmBN(xcb, 512, sw1b, B, 512, 512, s1, t1, h1b, 512);
    gemmBN(h1b, 512, sw2b, B, 256, 512, s2, t2, fusedb + 0, 768);

    // --- image encoders (shared weights), sequential to reuse pool buffers ---
    const int c1blocks = (B * 16 * 16 * 32) / 256;
    k_conv1<<<dim3(c1blocks), dim3(256), 0, stream>>>(x_img1, c1_w, sc1, tc1, poolA, B);
    k_conv2<<<dim3(B), dim3(256), 0, stream>>>(poolA, c2wb, sc2, tc2, poolB);
    gemmBN(poolB, 4096, ifcwb, B, 256, 4096, si, ti, fusedb + 256, 768);

    k_conv1<<<dim3(c1blocks), dim3(256), 0, stream>>>(x_img2, c1_w, sc1, tc1, poolA, B);
    k_conv2<<<dim3(B), dim3(256), 0, stream>>>(poolA, c2wb, sc2, tc2, poolB);
    gemmBN(poolB, 4096, ifcwb, B, 256, 4096, si, ti, fusedb + 512, 768);

    // --- router + aux ---
    k_gate<<<dim3(B / 256), dim3(256), 0, stream>>>(fusedb, gate_w, gate_b, probs, comb, B);
    k_aux<<<dim3(1), dim3(256), 0, stream>>>(probs, comb, (float*)d_out + (size_t)B * 12, B);

    // --- dense MoE (matches reference: all experts, top-2 combine) ---
    k_zero<<<dim3(512), dim3(256), 0, stream>>>(accF, (long)B * 256);
    for (int e = 0; e < 8; e++) {
        gemmB(fusedb, 768, ew1b + (size_t)e * 768 * 512, B, 512, 768,
              eb1 + e * 512, heb, 512);
        gemmEx2(heb, 512, ew2b + (size_t)e * 512 * 256, B, 256, 512,
                eb2 + e * 256, comb + e, 8, accF, 256);
    }

    // --- classifier ---
    cvt(accF, accb, (long)B * 256);
    gemmB(accb, 256, cw1b, B, 128, 256, cl_b1, hclb, 128);
    k_cls2<<<dim3((B * 12 + 255) / 256), dim3(256), 0, stream>>>(hclb, cl_w2, cl_b2, (float*)d_out, B);
}
